// SelfAttentionBlock_28303834480918
// MI455X (gfx1250) — matrix operand order repaired and checked by reading, not compiled, not run on hardware
//
#include <hip/hip_runtime.h>

// ---------------------------------------------------------------------------
// Problem constants (from reference): B=4, CX=CY=256, H*W=1024, K*h*w=8192, A=64
// ---------------------------------------------------------------------------
#define B_   4
#define CX_  256
#define HW_  1024
#define KK_  8
#define N_   8192
#define A_   64

typedef __attribute__((ext_vector_type(16))) __bf16 v16bf;
typedef __attribute__((ext_vector_type(8)))  __bf16 v8bf;
typedef __attribute__((ext_vector_type(8)))  float  v8f;

static __device__ __forceinline__ v8f wmma_bf16(v16bf a, v16bf b, v8f c) {
    // D = A(16x32 bf16) * B(32x16 bf16) + C(16x16 f32)
    return __builtin_amdgcn_wmma_f32_16x16x32_bf16(
        /*neg_a=*/false, a, /*neg_b=*/false, b,
        /*c_mod=*/(short)0, c, /*reuse_a=*/false, /*reuse_b=*/false);
}

static __device__ __forceinline__ v16bf cat8(v8bf lo, v8bf hi) {
    return __builtin_shufflevector(lo, hi, 0,1,2,3,4,5,6,7,8,9,10,11,12,13,14,15);
}

// ---------------------------------------------------------------------------
// zero scratch accumulators (O1 + rs are atomically accumulated each launch)
// ---------------------------------------------------------------------------
__global__ void zero_kernel(float* p, int n) {
    int i = blockIdx.x * 256 + threadIdx.x;
    if (i < n) p[i] = 0.0f;
}

// ---------------------------------------------------------------------------
// 2D sinusoidal PE value: Ppe[p][c], p = y*32+x, c in [0,256)
// ---------------------------------------------------------------------------
static __device__ __forceinline__ float pe_val(int p, int c) {
    int pos, ch;
    if (c < 128) { pos = p >> 5; ch = c; }          // pe_h(y), d=128
    else         { pos = p & 31; ch = c - 128; }    // pe_w(x), d=128
    float di  = __expf(-9.210340371976184f * (float)(ch & ~1) / 128.0f);
    float ang = (float)pos * di;
    return (ch & 1) ? __cosf(ang) : __sinf(ang);
}

// pq[p][a] = PE[p]*Wpx[a] + bpx[a] ; pk[p][a] = PE[p]*Wpy[a] + bpy[a]
__global__ void pe_proj_kernel(const float* __restrict__ Wpx, const float* __restrict__ bpx,
                               const float* __restrict__ Wpy, const float* __restrict__ bpy,
                               float* __restrict__ pq, float* __restrict__ pk) {
    int idx = blockIdx.x * 256 + threadIdx.x;   // 131072 = 2 * 1024 * 64
    int which = idx >> 16;
    int r = idx & 65535;
    int p = r >> 6, a = r & 63;
    const float* W = which ? Wpy : Wpx;
    float acc = which ? bpy[a] : bpx[a];
    for (int c = 0; c < 256; ++c) acc += pe_val(p, c) * W[a * 256 + c];
    (which ? pk : pq)[p * 64 + a] = acc;
}

// qb[b][p][a] = bf16( (zx_c[b,p]*Wq[a] + bq[a] + pq[p][a]) * 1/sqrt(64) )
__global__ void qproj_kernel(const float* __restrict__ zx, const float* __restrict__ Wq,
                             const float* __restrict__ bq, const float* __restrict__ pq,
                             __bf16* __restrict__ qb) {
    int g = blockIdx.x * 256 + threadIdx.x;     // 262144 = 4 * 1024 * 64
    int b = g >> 16; int r = g & 65535;
    int p = r >> 6, a = r & 63;
    float acc = bq[a] + pq[p * 64 + a];
    const float* zp = zx + (size_t)b * CX_ * HW_ + p;
    const float* wp = Wq + a * 256;
    for (int c = 0; c < 256; ++c) acc += zp[(size_t)c * HW_] * wp[c];
    qb[((size_t)b * HW_ + p) * 64 + a] = (__bf16)(acc * 0.125f);
}

// kb[b][n][a] = bf16( zy_c[b,n]*Wk[a] + bk[a] + pk[pos][a] ), n = kk*1024 + pos
__global__ void kproj_kernel(const float* __restrict__ zy, const float* __restrict__ Wk,
                             const float* __restrict__ bk, const float* __restrict__ pk,
                             __bf16* __restrict__ kb) {
    int g = blockIdx.x * 256 + threadIdx.x;     // 2097152 = 4 * 8192 * 64
    int b = g >> 19; int r = g & 0x7FFFF;
    int n = r >> 6, a = r & 63;
    int kk = n >> 10, pos = n & 1023;
    float acc = bk[a] + pk[pos * 64 + a];
    const float* zp = zy + (size_t)(b * KK_ + kk) * 256 * 1024 + pos;
    const float* wp = Wk + a * 256;
    for (int c = 0; c < 256; ++c) acc += zp[(size_t)c * 1024] * wp[c];
    kb[((size_t)b * N_ + n) * 64 + a] = (__bf16)acc;
}

// zyb = bf16(zy), same layout [B][K][CY][pos] so pos stays contiguous (B-fragment dim)
__global__ void cvt_kernel(const float* __restrict__ zy, __bf16* __restrict__ zyb, int n4) {
    int i = blockIdx.x * 256 + threadIdx.x;
    if (i < n4) {
        float4 v = ((const float4*)zy)[i];
        __bf16* o = zyb + (size_t)i * 4;
        o[0] = (__bf16)v.x; o[1] = (__bf16)v.y; o[2] = (__bf16)v.z; o[3] = (__bf16)v.w;
    }
}

// ---------------------------------------------------------------------------
// Async copy of one K-tile (32 k-rows x 64 a, all 4 batches = 16KB) into LDS
// via GLOBAL_LOAD_ASYNC_TO_LDS_B64: 8 chunks of 8B per thread, fully
// coalesced in global (kb rows are contiguous). Tracked with ASYNCcnt.
// ---------------------------------------------------------------------------
static __device__ __forceinline__ void issue_ktile(const __bf16* __restrict__ kb,
                                                   int kbase,
                                                   const __bf16* lbase, int tid) {
    #pragma unroll
    for (int m = 0; m < 8; ++m) {
        int q  = m * 256 + tid;     // chunk id in [0,2048)
        int bb = q >> 9;            // batch
        int r  = (q >> 4) & 31;     // k-row within tile
        int ag = q & 15;            // a-group of 4 elements (8B)
        const __bf16* g = kb + ((size_t)bb * N_ + kbase + r) * 64 + ag * 4;
        unsigned l = (unsigned)(unsigned long long)(lbase + ((size_t)bb * 32 + r) * 72 + ag * 4);
        asm volatile("global_load_async_to_lds_b64 %0, %1, off"
                     :: "v"(l), "v"(g) : "memory");
    }
}

static __device__ __forceinline__ void wait_async0() {
    asm volatile("s_wait_asynccnt 0" ::: "memory");
}

// ---------------------------------------------------------------------------
// Fused attention: S = Q K^T, softmax over BATCH axis, O += P * Zy_c.
// 512 blocks = 32 q-tiles x 16 k-splits, 256 threads = 8 waves,
// wave -> (batch b = w&3, q-halftile qh = w>>2).
// K-tiles double-buffered in LDS via async copies; zy B-fragments direct
// from L2-resident zyb.
// ---------------------------------------------------------------------------
__global__ __launch_bounds__(256) void attn_kernel(const __bf16* __restrict__ qb,
                                                   const __bf16* __restrict__ kb,
                                                   const __bf16* __restrict__ zyb,
                                                   float* __restrict__ O1,
                                                   float* __restrict__ rs) {
    __shared__ __attribute__((aligned(16))) __bf16 Kt[2][4][32][72];   // 36,864 B
    __shared__ float  Sbuf[2][4][16][34];                              // 17,408 B
    __shared__ __attribute__((aligned(16))) __bf16 Pbuf[2][4][16][40]; // 10,240 B
    __shared__ float  rsbuf[2][4][16];                                 //    512 B

    int qt = blockIdx.x >> 4;        // q tile   [0,32)
    int ks = blockIdx.x & 15;        // k split  [0,16)
    int tid = threadIdx.x;
    int wave = tid >> 5, lane = tid & 31;
    int b = wave & 3, qh = wave >> 2;
    int ln = lane & 15, hi = lane >> 4;

    // prologue: start async copy of K-tile 0, init rowsum partials
    issue_ktile(kb, ks * 512, &Kt[0][0][0][0], tid);
    if (tid < 128) rsbuf[tid >> 6][(tid >> 4) & 3][tid & 15] = 0.0f;

    // --- Q A-fragments (16 rows x 64 a), loaded once ---
    int qrow = qt * 32 + qh * 16 + ln;
    const __bf16* qp = qb + ((size_t)b * HW_ + qrow) * 64;
    v16bf Qf0 = cat8(*(const v8bf*)(qp +  0 + hi * 16), *(const v8bf*)(qp + 8 + hi * 16));
    v16bf Qf1 = cat8(*(const v8bf*)(qp + 32 + hi * 16), *(const v8bf*)(qp + 40 + hi * 16));

    v8f O[16];
    #pragma unroll
    for (int i = 0; i < 16; ++i)
        #pragma unroll
        for (int j = 0; j < 8; ++j) O[i][j] = 0.0f;

    wait_async0();          // K-tile 0 landed (this wave's chunks)
    __syncthreads();        // ... and everyone else's

    for (int it = 0; it < 16; ++it) {
        int kbase = ks * 512 + it * 32;       // global k index of this 32-chunk

        // ---- S(16x32) = Q(16x64) @ K^T from LDS K-tile ----
        #pragma unroll
        for (int nt = 0; nt < 2; ++nt) {
            const __bf16* kp = &Kt[it & 1][b][0][0] + (nt * 16 + ln) * 72;
            v16bf Bf0 = cat8(*(const v8bf*)(kp +      hi * 16),
                             *(const v8bf*)(kp +  8 + hi * 16));
            v16bf Bf1 = cat8(*(const v8bf*)(kp + 32 + hi * 16),
                             *(const v8bf*)(kp + 40 + hi * 16));
            v8f s;
            #pragma unroll
            for (int j = 0; j < 8; ++j) s[j] = 0.0f;
            s = wmma_bf16(Qf0, Bf0, s);
            s = wmma_bf16(Qf1, Bf1, s);
            #pragma unroll
            for (int i = 0; i < 8; ++i)
                Sbuf[qh][b][i + hi * 8][nt * 16 + ln] = s[i];
        }
        __syncthreads();

        // start async copy of the NEXT K-tile; overlaps the softmax phase.
        // Safe: all waves passed the barrier above, so nobody is still
        // reading buffer (it+1)&1 (last read was S-phase of iter it-1).
        if (it < 15)
            issue_ktile(kb, ks * 512 + (it + 1) * 32, &Kt[(it + 1) & 1][0][0][0], tid);

        // ---- softmax across the 4 batches (per (q,k) position) ----
        {
            int g0 = tid * 4;
            int sqh = g0 >> 9, rr = (g0 >> 5) & 15, cb = g0 & 31;
            float ls0 = 0.f, ls1 = 0.f, ls2 = 0.f, ls3 = 0.f;
            #pragma unroll
            for (int j = 0; j < 4; ++j) {
                int cc = cb + j;
                float s0 = Sbuf[sqh][0][rr][cc];
                float s1 = Sbuf[sqh][1][rr][cc];
                float s2 = Sbuf[sqh][2][rr][cc];
                float s3 = Sbuf[sqh][3][rr][cc];
                float m  = fmaxf(fmaxf(s0, s1), fmaxf(s2, s3));
                float e0 = __expf(s0 - m), e1 = __expf(s1 - m);
                float e2 = __expf(s2 - m), e3 = __expf(s3 - m);
                float inv = 1.0f / (e0 + e1 + e2 + e3);
                e0 *= inv; e1 *= inv; e2 *= inv; e3 *= inv;
                Pbuf[sqh][0][rr][cc] = (__bf16)e0;
                Pbuf[sqh][1][rr][cc] = (__bf16)e1;
                Pbuf[sqh][2][rr][cc] = (__bf16)e2;
                Pbuf[sqh][3][rr][cc] = (__bf16)e3;
                ls0 += e0; ls1 += e1; ls2 += e2; ls3 += e3;
            }
            atomicAdd(&rsbuf[sqh][0][rr], ls0);
            atomicAdd(&rsbuf[sqh][1][rr], ls1);
            atomicAdd(&rsbuf[sqh][2][rr], ls2);
            atomicAdd(&rsbuf[sqh][3][rr], ls3);
        }
        wait_async0();      // next K-tile landed (must precede the barrier
                            // so every wave's chunks are visible after it)
        __syncthreads();

        // ---- O(16x256) += P(16x32) @ Zy(32x256) ----
        const __bf16* pr = &Pbuf[qh][b][ln][0];
        v16bf Pf = cat8(*(const v8bf*)(pr + hi * 16), *(const v8bf*)(pr + 8 + hi * 16));
        int kkq  = kbase >> 10;        // which K slot of zy
        int posb = kbase & 1023;       // pos base inside slot (chunk never crosses)
        const __bf16* vbase = zyb + (size_t)(b * KK_ + kkq) * 256 * 1024 + posb + hi * 16;
        #pragma unroll
        for (int ct = 0; ct < 16; ++ct) {
            const __bf16* vp = vbase + (size_t)(ct * 16 + ln) * 1024;
            v16bf Vf = cat8(*(const v8bf*)vp, *(const v8bf*)(vp + 8));
            O[ct] = wmma_bf16(Pf, Vf, O[ct]);
        }
        // next iteration's Pbuf writes are fenced by the barrier after the S store
    }

    // ---- split-k combine: O1[b][q][c] += O ----
    int qb0 = qt * 32 + qh * 16;
    #pragma unroll
    for (int ct = 0; ct < 16; ++ct)
        #pragma unroll
        for (int i = 0; i < 8; ++i) {
            float* dst = O1 + ((size_t)b * HW_ + qb0 + i + hi * 8) * 256 + ct * 16 + ln;
            atomicAdd(dst, O[ct][i]);
        }

    __syncthreads();
    if (tid < 128) {
        int q2 = tid >> 6, b2 = (tid >> 4) & 3, r2 = tid & 15;
        atomicAdd(rs + (size_t)b2 * HW_ + qt * 32 + q2 * 16 + r2, rsbuf[q2][b2][r2]);
    }
}

// ---------------------------------------------------------------------------
// out[row][c] = O1[row][:] . Wv[c][:] + rs[row]*bv[c]   (rows flat over b,q)
// WMMA with on-the-fly f32->bf16 fragment conversion.
// ---------------------------------------------------------------------------
__global__ __launch_bounds__(256) void final_kernel(const float* __restrict__ O1,
                                                    const float* __restrict__ rs,
                                                    const float* __restrict__ Wv,
                                                    const float* __restrict__ bv,
                                                    float* __restrict__ out) {
    int tid = threadIdx.x, wave = tid >> 5, lane = tid & 31;
    int ln = lane & 15, hi = lane >> 4;
    int tile = blockIdx.x * 8 + wave;          // 4096 tiles: 256 row-tiles x 16 c-tiles
    int rowtile = tile >> 4, ct = tile & 15;
    int rowbase = rowtile * 16, cbase = ct * 16;

    const float* arow = O1 + (size_t)(rowbase + ln) * 256;   // A row for this lane
    const float* brow = Wv + (size_t)(cbase + ln) * 256;     // Wv row = output col c

    v8f acc;
    #pragma unroll
    for (int j = 0; j < 8; ++j) acc[j] = 0.0f;

    #pragma unroll
    for (int ch = 0; ch < 8; ++ch) {
        int cp = ch * 32;                                     // reduction chunk over c'
        const float* a0 = arow + cp + hi * 16;                // K = cp+hi*16  .. +7
        const float* a1 = arow + cp + 8 + hi * 16;            // K = cp+8+hi*16.. +7
        v16bf Af;
        #pragma unroll
        for (int j = 0; j < 8; ++j) { Af[j] = (__bf16)a0[j]; Af[8 + j] = (__bf16)a1[j]; }
        const float* bsrc = brow + cp + hi * 16;              // K = cp+hi*16 .. +15
        v16bf Bf;
        #pragma unroll
        for (int j = 0; j < 16; ++j) Bf[j] = (__bf16)bsrc[j];
        acc = wmma_bf16(Af, Bf, acc);
    }

    #pragma unroll
    for (int i = 0; i < 8; ++i) {
        int r = rowbase + i + hi * 8;
        int c = cbase + ln;
        out[(size_t)r * 256 + c] = acc[i] + rs[r] * bv[c];
    }
}

// ---------------------------------------------------------------------------
// Host launcher. Workspace layout (all 256B aligned, ~25.1 MB total):
//   pq   f32 [1024][64]          @ 0
//   pk   f32 [1024][64]          @ 0x040000
//   qb   bf16 [4][1024][64]      @ 0x080000
//   kb   bf16 [4][8192][64]      @ 0x100000
//   zyb  bf16 [4][8][256][1024]  @ 0x500000
//   O1   f32 [4][1024][256]      @ 0x1500000
//   rs   f32 [4][1024]           @ 0x1900000
// ---------------------------------------------------------------------------
extern "C" void kernel_launch(void* const* d_in, const int* in_sizes, int n_in,
                              void* d_out, int out_size, void* d_ws, size_t ws_size,
                              hipStream_t stream) {
    (void)in_sizes; (void)n_in; (void)out_size; (void)ws_size;

    const float* zx  = (const float*)d_in[0];
    const float* zy  = (const float*)d_in[1];
    const float* Wq  = (const float*)d_in[2];
    const float* bq  = (const float*)d_in[3];
    const float* Wpx = (const float*)d_in[4];
    const float* bpx = (const float*)d_in[5];
    const float* Wk  = (const float*)d_in[6];
    const float* bk  = (const float*)d_in[7];
    const float* Wpy = (const float*)d_in[8];
    const float* bpy = (const float*)d_in[9];
    const float* Wv  = (const float*)d_in[10];
    const float* bv  = (const float*)d_in[11];
    float* out = (float*)d_out;

    char* ws = (char*)d_ws;
    float*  pq  = (float*)(ws + 0x000000);
    float*  pk  = (float*)(ws + 0x040000);
    __bf16* qb  = (__bf16*)(ws + 0x080000);
    __bf16* kb  = (__bf16*)(ws + 0x100000);
    __bf16* zyb = (__bf16*)(ws + 0x500000);
    float*  O1  = (float*)(ws + 0x1500000);
    float*  rs  = (float*)(ws + 0x1900000);

    // O1 (1,048,576) and rs (4,096) are contiguous: zero both.
    zero_kernel<<<4112, 256, 0, stream>>>(O1, 1048576 + 4096);
    pe_proj_kernel<<<512, 256, 0, stream>>>(Wpx, bpx, Wpy, bpy, pq, pk);
    qproj_kernel<<<1024, 256, 0, stream>>>(zx, Wq, bq, pq, qb);
    kproj_kernel<<<8192, 256, 0, stream>>>(zy, Wk, bk, pk, kb);
    cvt_kernel<<<8192, 256, 0, stream>>>(zy, zyb, (B_ * KK_ * 256 * 1024) / 4);
    attn_kernel<<<512, 256, 0, stream>>>(qb, kb, zyb, O1, rs);
    final_kernel<<<512, 256, 0, stream>>>(O1, rs, Wv, bv, out);
}
